// InhBlock_63565515981231
// MI455X (gfx1250) — matrix operand order repaired and checked by reading, not compiled, not run on hardware
//
#include <hip/hip_runtime.h>
#include <hip/hip_bf16.h>

// ---- problem constants (match reference) ----
#define BB   32
#define LL   12
#define NNODE 512
#define DD   128
#define HH   4
#define HDIM 32
#define FKO  256
#define TT   12                      // OUT_STEPS
#define BN   (BB*NNODE)              // 16384
#define LMAX (LL + TT - 1)           // 23
#define BND  ((size_t)BN*DD)         // 2,097,152 = 2^21
#define WPITCH 136                   // LDS row pitch in bf16 (272B = 17*16B)

typedef __attribute__((ext_vector_type(16))) __bf16 v16bf;
typedef __attribute__((ext_vector_type(8)))  float  v8f;

__device__ __forceinline__ __bf16 f2bf(float f) {
  unsigned int u = __builtin_bit_cast(unsigned int, f);
  u += 0x7FFFu + ((u >> 16) & 1u);                 // RNE to bf16
  unsigned short h = (unsigned short)(u >> 16);
  return __builtin_bit_cast(__bf16, h);
}

// =====================================================================
// GEMM: C[M, Nout] = A(M x 128, fp32, row-major) @ W^T + bias
// W is (Nout, 128) bf16 row-major in global. 256-thread block = 8 waves.
// Block stages the full weight into LDS once via CDNA5 async copies
// (global_load_async_to_lds_b128, ASYNCcnt), then each wave computes one
// 16-row M tile, sweeping all 16-col N tiles: K=128 -> 4x
// v_wmma_f32_16x16x32_bf16 per tile, B fragments from LDS (ds_load_b128).
// =====================================================================
__global__ __launch_bounds__(256) void gemm_bf16_wmma(
    const float* __restrict__ A,
    const __bf16* __restrict__ W,
    const float* __restrict__ bias,
    float* __restrict__ C, int ldc, int ntiles)
{
  extern __shared__ char smem[];
  __bf16* sW = (__bf16*)smem;                 // (ntiles*16) rows x WPITCH pitch

  const int tid = threadIdx.x;                // 0..255

  // ---- stage W -> LDS with async DMA-style copies (16B per op) ----
  const int nchunk = ntiles * 256;            // (ntiles*16 rows) * (128/8 chunks)
  for (int c = tid; c < nchunk; c += 256) {
    const int row = c >> 4;
    const int col = (c & 15) * 8;
    unsigned lds = (unsigned)(size_t)(sW + (size_t)row * WPITCH + col);
    unsigned long long g = (unsigned long long)(W + (size_t)row * DD + col);
    asm volatile("global_load_async_to_lds_b128 %0, %1, off"
                 :: "v"(lds), "v"(g) : "memory");
  }
  asm volatile("s_wait_asynccnt 0" ::: "memory");
  __syncthreads();

  // ---- per-wave 16xNout tile ----
  const int wave  = tid >> 5;
  const int lane  = tid & 31;
  const int mrow  = lane & 15;                // A row within tile (both halves)
  const int khalf = lane >> 4;                // K-half selector (ISA 16-bit A layout)
  const size_t mbase = (size_t)blockIdx.x * 128 + (size_t)wave * 16;

  const float* __restrict__ Arow = A + (mbase + (size_t)mrow) * DD;
  __builtin_prefetch((const void*)Arow, 0, 0);          // global_prefetch_b8
  v16bf afrag[4];
  #pragma unroll
  for (int ks = 0; ks < 4; ++ks) {
    #pragma unroll
    for (int e = 0; e < 16; ++e) {
      // 16-bit A 16x32: element e -> K = base + (e>=8?16:0) + khalf*8 + (e&7)
      int k = ks*32 + ((e >= 8) ? 16 : 0) + khalf*8 + (e & 7);
      afrag[ks][e] = f2bf(Arow[k]);
    }
  }

  for (int nt = 0; nt < ntiles; ++nt) {
    const int ncol = nt*16 + (lane & 15);
    const __bf16* __restrict__ wr = sW + (size_t)ncol * WPITCH;
    v8f c = {0.f,0.f,0.f,0.f,0.f,0.f,0.f,0.f};
    #pragma unroll
    for (int ks = 0; ks < 4; ++ks) {
      v16bf bfrag;
      #pragma unroll
      for (int e = 0; e < 16; ++e) {
        // 16-bit B 32x16: lane holds column N=lane&15, K = (e>=8?16:0) + khalf*8 + (e&7)
        bfrag[e] = wr[ks*32 + ((e >= 8) ? 16 : 0) + khalf*8 + (e & 7)];
      }
      c = __builtin_amdgcn_wmma_f32_16x16x32_bf16(
            false, afrag[ks], false, bfrag, (short)0, c, false, false);
    }
    const float bv = bias ? bias[ncol] : 0.0f;
    #pragma unroll
    for (int r = 0; r < 8; ++r) {
      // C/D layout: lanes 0-15 -> M=r, lanes 16-31 -> M=8+r; N = lane&15
      size_t row = mbase + (size_t)(khalf ? 8 + r : r);
      C[row * (size_t)ldc + ncol] = c[r] + bv;
    }
  }
}

// ---------------- small helper kernels ----------------
__global__ void fill_kernel(float* __restrict__ p, float v) {
  size_t i = (size_t)blockIdx.x * blockDim.x + threadIdx.x;
  p[i] = v;
}

__global__ void copy_kernel(float* __restrict__ dst, const float* __restrict__ src) {
  size_t i = (size_t)blockIdx.x * blockDim.x + threadIdx.x;
  dst[i] = src[i];
}

__global__ void f2bf_kernel(const float* __restrict__ src, __bf16* __restrict__ dst) {
  size_t i = (size_t)blockIdx.x * blockDim.x + threadIdx.x;
  dst[i] = f2bf(src[i]);
}

// sinusoidal PE table: pe[pos*128 + 2i] = sin(pos*div_i), +1 -> cos
__global__ void pe_kernel(float* __restrict__ pe) {
  const int pos = blockIdx.x;    // 0..63
  const int i   = threadIdx.x;   // 0..63
  const float div = __expf(-(float)(2*i) * (9.210340371976184f / 128.0f)); // ln(1e4)
  const float a = (float)pos * div;
  pe[pos*DD + 2*i]     = sinf(a);
  pe[pos*DD + 2*i + 1] = cosf(a);
}

// X (B,L,N,D) -> Xs (L, BN, D), BN index = b*N+n
__global__ void reorder_x_kernel(const float* __restrict__ X, float* __restrict__ Xs) {
  size_t i = (size_t)blockIdx.x * blockDim.x + threadIdx.x;   // dst linear
  int l = (int)(i >> 21);                 // / (BN*D)
  size_t r = i & (BND - 1);
  size_t bn = r >> 7;  int d = (int)(r & 127);
  size_t b = bn >> 9;  int n = (int)(bn & 511);
  Xs[i] = X[(((size_t)b*LL + l)*NNODE + n)*DD + d];
}

// dst[l][bn][d] = src[l][bn][d] + pe[l][d]   (in-place OK)
__global__ void addpe_kernel(const float* __restrict__ src, const float* __restrict__ pe,
                             float* __restrict__ dst) {
  size_t i = (size_t)blockIdx.x * blockDim.x + threadIdx.x;
  int l = (int)(i >> 21);
  int d = (int)(i & 127);
  dst[i] = src[i] + pe[l*DD + d];
}

// GRU elementwise: gi/gh are (BN,384) in [r|z|n] order; h is (BN,128)
__global__ void gru_elem_kernel(const float* __restrict__ gi, const float* __restrict__ gh,
                                const float* __restrict__ h,
                                float* __restrict__ hout, float* __restrict__ rnnh_row) {
  size_t i = (size_t)blockIdx.x * blockDim.x + threadIdx.x;   // over BN*D
  size_t bn = i >> 7;  int d = (int)(i & 127);
  const float* gi_r = gi + bn*384;
  const float* gh_r = gh + bn*384;
  float r  = 1.0f / (1.0f + __expf(-(gi_r[d]       + gh_r[d])));
  float z  = 1.0f / (1.0f + __expf(-(gi_r[128 + d] + gh_r[128 + d])));
  float n  = tanhf(gi_r[256 + d] + r * gh_r[256 + d]);
  float hv = h[i];
  float out = (1.0f - z) * n + z * hv;
  hout[i] = out;
  rnnh_row[i] = out;
}

// Full self-attention for Z0: qkv (L*BN, 384) = [q|k|v], ctx (L*BN, 128)
// one thread per (ql, bn, h)
__global__ __launch_bounds__(128) void attn_full_kernel(const float* __restrict__ qkv,
                                                        float* __restrict__ ctx) {
  const int tid = blockIdx.x * 128 + threadIdx.x;   // L*BN*H = 786432
  const int h = tid & 3;
  const size_t bn = (size_t)((tid >> 2) & (BN - 1));
  const int ql = tid >> 16;
  const float* qp = qkv + ((size_t)ql*BN + bn)*384 + h*HDIM;
  float q[HDIM];
  #pragma unroll
  for (int d = 0; d < HDIM; ++d) q[d] = qp[d];
  float s[LL]; float mx = -1e30f;
  for (int kl = 0; kl < LL; ++kl) {
    const float* kp = qkv + ((size_t)kl*BN + bn)*384 + 128 + h*HDIM;
    float acc = 0.f;
    #pragma unroll
    for (int d = 0; d < HDIM; ++d) acc += q[d] * kp[d];
    acc *= 0.17677669529663687f;   // 1/sqrt(32)
    s[kl] = acc; mx = fmaxf(mx, acc);
  }
  float sum = 0.f;
  for (int kl = 0; kl < LL; ++kl) { s[kl] = __expf(s[kl] - mx); sum += s[kl]; }
  const float inv = 1.0f / sum;
  float o[HDIM];
  #pragma unroll
  for (int d = 0; d < HDIM; ++d) o[d] = 0.f;
  for (int kl = 0; kl < LL; ++kl) {
    const float* vp = qkv + ((size_t)kl*BN + bn)*384 + 256 + h*HDIM;
    const float w = s[kl] * inv;
    #pragma unroll
    for (int d = 0; d < HDIM; ++d) o[d] += w * vp[d];
  }
  float* op = ctx + ((size_t)ql*BN + bn)*DD + h*HDIM;
  #pragma unroll
  for (int d = 0; d < HDIM; ++d) op[d] = o[d];
}

// Decode attention: qg (BN,128), kv (len*BN, 256) = [k|v], ctx (BN,128)
__global__ __launch_bounds__(128) void attn_dec_kernel(const float* __restrict__ qg,
                                                       const float* __restrict__ kv,
                                                       float* __restrict__ ctx, int len) {
  const int tid = blockIdx.x * 128 + threadIdx.x;   // BN*H = 65536
  const int h = tid & 3;
  const size_t bn = (size_t)(tid >> 2);
  const float* qp = qg + bn*DD + h*HDIM;
  float q[HDIM];
  #pragma unroll
  for (int d = 0; d < HDIM; ++d) q[d] = qp[d];
  float s[LMAX]; float mx = -1e30f;
  for (int kl = 0; kl < len; ++kl) {
    const float* kp = kv + ((size_t)kl*BN + bn)*256 + h*HDIM;
    float acc = 0.f;
    #pragma unroll
    for (int d = 0; d < HDIM; ++d) acc += q[d] * kp[d];
    acc *= 0.17677669529663687f;
    s[kl] = acc; mx = fmaxf(mx, acc);
  }
  float sum = 0.f;
  for (int kl = 0; kl < len; ++kl) { s[kl] = __expf(s[kl] - mx); sum += s[kl]; }
  const float inv = 1.0f / sum;
  float o[HDIM];
  #pragma unroll
  for (int d = 0; d < HDIM; ++d) o[d] = 0.f;
  for (int kl = 0; kl < len; ++kl) {
    const float* vp = kv + ((size_t)kl*BN + bn)*256 + 128 + h*HDIM;
    const float w = s[kl] * inv;
    #pragma unroll
    for (int d = 0; d < HDIM; ++d) o[d] += w * vp[d];
  }
  float* op = ctx + bn*DD + h*HDIM;
  #pragma unroll
  for (int d = 0; d < HDIM; ++d) op[d] = o[d];
}

// ResidualDecomp: res = LN(X - relu(back)); one 128-thread block per (b,l,n) row
__global__ __launch_bounds__(128) void backln_kernel(const float* __restrict__ X,
                                                     const float* __restrict__ back,
                                                     const float* __restrict__ gamma,
                                                     const float* __restrict__ beta,
                                                     float* __restrict__ res) {
  const size_t row = blockIdx.x;            // (b,l,n) order == X/res row order
  const int d = threadIdx.x;
  const size_t b = row / ((size_t)LL * NNODE);
  const int l = (int)((row / NNODE) % LL);
  const int n = (int)(row % NNODE);
  const float x  = X[row*DD + d];
  const float bk = back[(((size_t)l*BN) + b*NNODE + n)*DD + d];
  const float u  = x - fmaxf(bk, 0.f);
  __shared__ float s1[DD], s2[DD];
  s1[d] = u; s2[d] = u*u;
  __syncthreads();
  for (int off = 64; off > 0; off >>= 1) {
    if (d < off) { s1[d] += s1[d + off]; s2[d] += s2[d + off]; }
    __syncthreads();
  }
  const float mu  = s1[0] * (1.0f/128.0f);
  const float var = s2[0] * (1.0f/128.0f) - mu*mu;
  res[row*DD + d] = (u - mu) * rsqrtf(var + 1e-5f) * gamma[d] + beta[d];
}

// fcbuf (T*BN, 256) [row = t*BN + b*N + n] -> out (B,T,N,FK)
__global__ void reorder_fc_kernel(const float* __restrict__ fc, float* __restrict__ out) {
  size_t i = (size_t)blockIdx.x * blockDim.x + threadIdx.x;
  int f = (int)(i & 255);
  size_t r = i >> 8;
  size_t bn = r & (BN - 1);
  int t = (int)(r >> 14);
  size_t b = bn >> 9;  int n = (int)(bn & 511);
  out[(((size_t)b*TT + t)*NNODE + n)*FKO + f] = fc[i];
}

// =====================================================================
extern "C" void kernel_launch(void* const* d_in, const int* in_sizes, int n_in,
                              void* d_out, int out_size, void* d_ws, size_t ws_size,
                              hipStream_t stream) {
  (void)in_sizes; (void)n_in; (void)out_size; (void)ws_size;

  const float* X         = (const float*)d_in[0];
  const float* gru_w_ih  = (const float*)d_in[1];
  const float* gru_w_hh  = (const float*)d_in[2];
  const float* gru_b_ih  = (const float*)d_in[3];
  const float* gru_b_hh  = (const float*)d_in[4];
  const float* attn_in_w = (const float*)d_in[5];
  const float* attn_in_b = (const float*)d_in[6];
  const float* attn_out_w= (const float*)d_in[7];
  const float* attn_out_b= (const float*)d_in[8];
  const float* backcast_w= (const float*)d_in[9];
  const float* backcast_b= (const float*)d_in[10];
  const float* forecast_w= (const float*)d_in[11];
  const float* forecast_b= (const float*)d_in[12];
  const float* ln_gamma  = (const float*)d_in[13];
  const float* ln_beta   = (const float*)d_in[14];

  float* out_res = (float*)d_out;                       // (B,L,N,D)
  float* out_fc  = out_res + (size_t)BB*LL*NNODE*DD;    // (B,T,N,FK)

  // ---- workspace carve-up (256B aligned chunks) ----
  float* ws = (float*)d_ws;
  size_t off = 0;
  auto take = [&](size_t nfloats) -> float* {
    float* p = ws + off;
    off += (nfloats + 63) & ~(size_t)63;
    return p;
  };
  float*  PE   = take(64*DD);
  float*  WBFf = take(106496);                   // 212,992 bf16
  __bf16* WBF  = (__bf16*)WBFf;
  __bf16* bw_gih  = WBF;                         // (384,128)
  __bf16* bw_ghh  = WBF + 49152;                 // (384,128)
  __bf16* bw_attn = WBF + 98304;                 // (384,128) [q|k|v rows]
  __bf16* bw_outp = WBF + 147456;                // (128,128)
  __bf16* bw_back = WBF + 163840;                // (128,128)
  __bf16* bw_fc   = WBF + 180224;                // (256,128)

  float* Hb   = take(BND);                       // GRU hidden
  float* GI   = take((size_t)BN*384);
  float* GH   = take((size_t)BN*384);
  float* GB   = take(BND);                       // decode g (pre-PE)
  float* QG   = take(BND);                       // decode q projection
  float* XS   = take((size_t)LL*BND);            // Xs / rnn_pe / backcast (aliased phases)
  float* RNNH = take((size_t)LMAX*BND);          // growing RNN_H
  float* BIGB = take((size_t)LMAX*BN*256);       // Z0 qkv / decode kv / fcbuf (aliased)
  float* CTX  = take((size_t)LL*BND);            // attention context
  float* Z0   = take((size_t)LL*BND);
  float* PRED = take((size_t)TT*BND);

  auto gemm = [&](const float* A, const __bf16* W, const float* bias,
                  float* C, int ldc, int M, int Nout) {
    const int ntiles = Nout / 16;
    const size_t shmem = (size_t)ntiles * 16 * WPITCH * sizeof(__bf16);
    gemm_bf16_wmma<<<dim3(M/128), dim3(256), shmem, stream>>>(A, W, bias, C, ldc, ntiles);
  };

  // 0) PE table + bf16 weight conversion
  pe_kernel<<<dim3(64), dim3(64), 0, stream>>>(PE);
  f2bf_kernel<<<dim3(49152/256), dim3(256), 0, stream>>>(gru_w_ih,  bw_gih);
  f2bf_kernel<<<dim3(49152/256), dim3(256), 0, stream>>>(gru_w_hh,  bw_ghh);
  f2bf_kernel<<<dim3(49152/256), dim3(256), 0, stream>>>(attn_in_w, bw_attn);
  f2bf_kernel<<<dim3(16384/256), dim3(256), 0, stream>>>(attn_out_w, bw_outp);
  f2bf_kernel<<<dim3(16384/256), dim3(256), 0, stream>>>(backcast_w, bw_back);
  f2bf_kernel<<<dim3(32768/256), dim3(256), 0, stream>>>(forecast_w, bw_fc);

  // 1) X -> (L,BN,D); h0 = 0
  reorder_x_kernel<<<dim3((unsigned)(LL*BND/256)), dim3(256), 0, stream>>>(X, XS);
  fill_kernel<<<dim3((unsigned)(BND/256)), dim3(256), 0, stream>>>(Hb, 0.0f);

  // 2) GRU scan over L steps -> RNNH rows 0..11 (rnn_raw, no PE)
  for (int l = 0; l < LL; ++l) {
    gemm(XS + (size_t)l*BND, bw_gih, gru_b_ih, GI, 384, BN, 384);
    gemm(Hb,                 bw_ghh, gru_b_hh, GH, 384, BN, 384);
    gru_elem_kernel<<<dim3((unsigned)(BND/256)), dim3(256), 0, stream>>>(
        GI, GH, Hb, Hb, RNNH + (size_t)l*BND);
  }

  // 3) rnn_pe = rnn_raw + pe (non-mutating); QKV; self-attention; out proj -> Z0
  addpe_kernel<<<dim3((unsigned)(LL*BND/256)), dim3(256), 0, stream>>>(RNNH, PE, XS);
  gemm(XS, bw_attn, attn_in_b, BIGB, 384, LL*BN, 384);            // qkv (L*BN,384)
  attn_full_kernel<<<dim3((LL*BN*HH)/128), dim3(128), 0, stream>>>(BIGB, CTX);
  gemm(CTX, bw_outp, attn_out_b, Z0, 128, LL*BN, 128);

  // 4) predict[0] = Z0[-1]
  copy_kernel<<<dim3((unsigned)(BND/256)), dim3(256), 0, stream>>>(
      PRED, Z0 + (size_t)(LL-1)*BND);

  // 5) decode loop with growing KV (PE re-added cumulatively, as in source)
  for (int t = 1; t < TT; ++t) {
    const int len_prev = LL + (t - 1);
    const float* xprev = PRED + (size_t)(t-1)*BND;
    const float* hprev = RNNH + (size_t)(len_prev-1)*BND;
    gemm(xprev, bw_gih, gru_b_ih, GI, 384, BN, 384);
    gemm(hprev, bw_ghh, gru_b_hh, GH, 384, BN, 384);
    gru_elem_kernel<<<dim3((unsigned)(BND/256)), dim3(256), 0, stream>>>(
        GI, GH, hprev, GB, RNNH + (size_t)len_prev*BND);          // append g
    const int len = len_prev + 1;
    addpe_kernel<<<dim3((unsigned)((size_t)len*BND/256)), dim3(256), 0, stream>>>(
        RNNH, PE, RNNH);                                          // RNN_H += pe (in place)
    gemm(GB, bw_attn, attn_in_b, QG, 128, BN, 128);               // q from g (pre-PE)
    gemm(RNNH, bw_attn + (size_t)128*DD, attn_in_b + 128,
         BIGB, 256, len*BN, 256);                                 // k|v over full RNN_H
    attn_dec_kernel<<<dim3((BN*HH)/128), dim3(128), 0, stream>>>(QG, BIGB, CTX, len);
    gemm(CTX, bw_outp, attn_out_b, PRED + (size_t)t*BND, 128, BN, 128);
  }

  // 6) backcast + residual LayerNorm -> out_res
  gemm(Z0, bw_back, backcast_b, XS, 128, LL*BN, 128);             // back (L*BN,128)
  backln_kernel<<<dim3((unsigned)((size_t)BB*LL*NNODE)), dim3(128), 0, stream>>>(
      X, XS, ln_gamma, ln_beta, out_res);

  // 7) forecast head -> out_fc
  gemm(PRED, bw_fc, forecast_b, BIGB, 256, TT*BN, 256);           // (T*BN,256)
  reorder_fc_kernel<<<dim3((unsigned)((size_t)TT*BN*FKO/256)), dim3(256), 0, stream>>>(
      BIGB, out_fc);
}
